// MS_SSA_Conv_9990093930856
// MI455X (gfx1250) — matrix operand order repaired and checked by reading, not compiled, not run on hardware
//
#include <hip/hip_runtime.h>
#include <hip/hip_bf16.h>

typedef _Float16 half_t;
typedef __attribute__((ext_vector_type(16))) _Float16 v16h;
typedef __attribute__((ext_vector_type(8)))  _Float16 v8h;
typedef __attribute__((ext_vector_type(8)))  float    v8f;

#define T_STEPS 4
#define BATCH   32
#define CH      512
#define HW      196
#define MROWS   (BATCH*HW)        // 6272 rows per time step
#define MTOT    (T_STEPS*MROWS)   // 25088 total GEMM rows
#define LDA     40                // padded LDS row stride (halves) -> conflict-free b128 reads
#define LDB     40

// ---------------------------------------------------------------------------
// Weights fp32 -> f16 (spike activations are exactly representable in f16;
// only weights round, which is the standard precision tradeoff for WMMA).
// ---------------------------------------------------------------------------
__global__ __launch_bounds__(256) void k_wcvt(const float* __restrict__ a,
                                              const float* __restrict__ b,
                                              const float* __restrict__ c,
                                              const float* __restrict__ d,
                                              half_t* __restrict__ Wh) {
    int i = blockIdx.x * blockDim.x + threadIdx.x;   // 0..CH*CH-1
    Wh[i]                 = (half_t)a[i];
    Wh[CH * CH + i]       = (half_t)b[i];
    Wh[2 * CH * CH + i]   = (half_t)c[i];
    Wh[3 * CH * CH + i]   = (half_t)d[i];
}

// ---------------------------------------------------------------------------
// shortcut LIF on x: sequential over T, hard reset, v_th = 1.
// Output: binary spikes as f16 in GEMM layout Xs[t][b*HW+n][c].
// ---------------------------------------------------------------------------
__global__ __launch_bounds__(256) void k_lif_x(const float* __restrict__ x,
                                               half_t* __restrict__ Xs) {
    int idx = blockIdx.x * blockDim.x + threadIdx.x;  // 0 .. MROWS*CH-1
    int c   = idx & (CH - 1);
    int row = idx >> 9;                               // b*HW + n
    int b   = row / HW;
    int n   = row - b * HW;
    float v = 0.0f;
#pragma unroll
    for (int t = 0; t < T_STEPS; ++t) {
        float xv = x[((size_t)((t * BATCH + b) * CH + c)) * HW + n];
        v = 0.5f * (v + xv);                 // v += (x - v)/tau, tau = 2
        float s = (v >= 1.0f) ? 1.0f : 0.0f;
        Xs[((size_t)t * MROWS + row) * CH + c] = (half_t)s;
        v = (v >= 1.0f) ? 0.0f : v;          // hard reset
    }
}

// ---------------------------------------------------------------------------
// Fused qkv GEMM + BN + LIF.  grid = (MROWS/128, CH/64, 3 matrices).
// 8 waves / block; each wave computes a 32x32 tile via 2x2 WMMA 16x16x32 f16.
// The t-loop is inside: LIF membrane state stays in registers (C/D layout).
// ---------------------------------------------------------------------------
__global__ __launch_bounds__(256) void k_qkv_gemm_bn_lif(
    const half_t* __restrict__ Xs, const half_t* __restrict__ Wh,
    const float* __restrict__ bn,
    half_t* __restrict__ Sq, half_t* __restrict__ Sk, half_t* __restrict__ Sv) {
    __shared__ __attribute__((aligned(16))) half_t As[128 * LDA];
    __shared__ __attribute__((aligned(16))) half_t Bs[64 * LDB];

    const int mat = blockIdx.z;
    const half_t* W   = Wh + (size_t)mat * CH * CH;
    const float*  bnp = bn + mat * 4 * CH;
    half_t* Sout = (mat == 0) ? Sq : (mat == 1 ? Sk : Sv);

    const int row0 = blockIdx.x * 128;   // within one time step's MROWS
    const int col0 = blockIdx.y * 64;
    const int tid  = threadIdx.x;
    const int lane = tid & 31;
    const int wave = tid >> 5;
    const int wr   = (wave & 3) * 32;    // wave row offset inside tile
    const int wc   = (wave >> 2) * 32;   // wave col offset inside tile
    const int lrow = lane & 15;
    const int hi   = lane >> 4;          // lane group (ISA fragment layout)

    v8f state[2][2] = {};                // LIF membrane across t (C/D layout)

    for (int t = 0; t < T_STEPS; ++t) {
        v8f acc[2][2] = {};
        const half_t* Abase = Xs + ((size_t)t * MROWS + row0) * CH;

        for (int kb = 0; kb < CH; kb += 32) {
            {   // stage A tile: 128 rows x 32 halves
                int r = tid >> 1;
                int h = (tid & 1) * 16;
                const half_t* s = Abase + (size_t)r * CH + kb + h;
                *(v8h*)&As[r * LDA + h]     = *(const v8h*)(s);
                *(v8h*)&As[r * LDA + h + 8] = *(const v8h*)(s + 8);
            }
            {   // stage B tile (weight rows): 64 x 32 halves
                int r = tid >> 2;
                int h = (tid & 3) * 8;
                const half_t* s = W + (size_t)(col0 + r) * CH + kb + h;
                *(v8h*)&Bs[r * LDB + h] = *(const v8h*)(s);
            }
            __syncthreads();

            v16h af[2], bf[2];
#pragma unroll
            for (int i = 0; i < 2; ++i) {
                // A lane layout: row lrow, K {16*hi .. 16*hi+15}
                const half_t* ap = &As[(wr + i * 16 + lrow) * LDA + hi * 16];
                v8h alo = *(const v8h*)ap;
                v8h ahi = *(const v8h*)(ap + 8);
                af[i] = __builtin_shufflevector(alo, ahi, 0, 1, 2, 3, 4, 5, 6, 7,
                                                8, 9, 10, 11, 12, 13, 14, 15);
                // B lane layout: col lrow, K {16*hi .. 16*hi+15}
                const half_t* bp = &Bs[(wc + i * 16 + lrow) * LDB + hi * 16];
                v8h blo = *(const v8h*)bp;
                v8h bhi = *(const v8h*)(bp + 8);
                bf[i] = __builtin_shufflevector(blo, bhi, 0, 1, 2, 3, 4, 5, 6, 7,
                                                8, 9, 10, 11, 12, 13, 14, 15);
            }
#pragma unroll
            for (int i = 0; i < 2; ++i)
#pragma unroll
                for (int j = 0; j < 2; ++j)
                    acc[i][j] = __builtin_amdgcn_wmma_f32_16x16x32_f16(
                        false, af[i], false, bf[j], (short)0, acc[i][j], false, false);
            __syncthreads();
        }

        // Epilogue: BN + LIF, emit binary spikes. D layout: VGPR r -> M=r+8*hi, N=lrow.
#pragma unroll
        for (int j = 0; j < 2; ++j) {
            int o = col0 + wc + j * 16 + lrow;
            float g  = bnp[o];
            float be = bnp[CH + o];
            float mn = bnp[2 * CH + o];
            float vr = bnp[3 * CH + o];
            float sc = g * rsqrtf(vr + 1e-5f);
#pragma unroll
            for (int i = 0; i < 2; ++i) {
                int mbase = row0 + wr + i * 16 + hi * 8;
#pragma unroll
                for (int r = 0; r < 8; ++r) {
                    float y = (acc[i][j][r] - mn) * sc + be;
                    float v = 0.5f * (state[i][j][r] + y);
                    float s = (v >= 1.0f) ? 1.0f : 0.0f;
                    state[i][j][r] = (v >= 1.0f) ? 0.0f : v;
                    Sout[((size_t)t * MROWS + mbase + r) * CH + o] = (half_t)s;
                }
            }
        }
    }
}

// ---------------------------------------------------------------------------
// kv = sum over N of k*v per (t,b,channel), then LIF with v_th = 0.5.
// ---------------------------------------------------------------------------
__global__ __launch_bounds__(256) void k_kv_lif(const half_t* __restrict__ Ks,
                                                const half_t* __restrict__ Vs,
                                                half_t* __restrict__ kvs) {
    int idx = blockIdx.x * blockDim.x + threadIdx.x;  // 0 .. BATCH*CH-1
    int c = idx & (CH - 1);
    int b = idx >> 9;
    float v = 0.0f;
    for (int t = 0; t < T_STEPS; ++t) {
        size_t base = ((size_t)t * MROWS + (size_t)b * HW) * CH + c;
        float sum = 0.0f;
        for (int n = 0; n < HW; ++n)
            sum += (float)Ks[base + (size_t)n * CH] * (float)Vs[base + (size_t)n * CH];
        v = 0.5f * (v + sum);
        float s = (v >= 0.5f) ? 1.0f : 0.0f;
        kvs[(t * BATCH + b) * CH + c] = (half_t)s;
        v = (v >= 0.5f) ? 0.0f : v;
    }
}

// ---------------------------------------------------------------------------
// out = q * kv (binary x binary -> binary), broadcast over N. Writes the
// projection-GEMM input (aliases the Xs buffer, which is dead by now).
// ---------------------------------------------------------------------------
__global__ __launch_bounds__(256) void k_mul_q_kv(const half_t* __restrict__ Qs,
                                                  const half_t* __restrict__ kvs,
                                                  half_t* __restrict__ P) {
    size_t idx = (size_t)blockIdx.x * blockDim.x + threadIdx.x;  // 0..MTOT*CH-1
    int c  = (int)(idx & (CH - 1));
    int gr = (int)(idx >> 9);
    int t  = gr / MROWS;
    int rr = gr - t * MROWS;
    int b  = rr / HW;
    P[idx] = (half_t)((float)Qs[idx] * (float)kvs[(t * BATCH + b) * CH + c]);
}

// ---------------------------------------------------------------------------
// Projection GEMM + bias + BN + residual. Writes directly in (T,B,C,H,W).
// ---------------------------------------------------------------------------
__global__ __launch_bounds__(256) void k_proj_gemm_bn_add(
    const half_t* __restrict__ Ain, const half_t* __restrict__ Wp,
    const float* __restrict__ bproj, const float* __restrict__ bnp,
    const float* __restrict__ xid, float* __restrict__ out) {
    __shared__ __attribute__((aligned(16))) half_t As[128 * LDA];
    __shared__ __attribute__((aligned(16))) half_t Bs[64 * LDB];

    const int row0 = blockIdx.x * 128;   // global row within MTOT
    const int col0 = blockIdx.y * 64;
    const int tid  = threadIdx.x;
    const int lane = tid & 31;
    const int wave = tid >> 5;
    const int wr   = (wave & 3) * 32;
    const int wc   = (wave >> 2) * 32;
    const int lrow = lane & 15;
    const int hi   = lane >> 4;

    v8f acc[2][2] = {};
    const half_t* Abase = Ain + (size_t)row0 * CH;

    for (int kb = 0; kb < CH; kb += 32) {
        {
            int r = tid >> 1;
            int h = (tid & 1) * 16;
            const half_t* s = Abase + (size_t)r * CH + kb + h;
            *(v8h*)&As[r * LDA + h]     = *(const v8h*)(s);
            *(v8h*)&As[r * LDA + h + 8] = *(const v8h*)(s + 8);
        }
        {
            int r = tid >> 2;
            int h = (tid & 3) * 8;
            const half_t* s = Wp + (size_t)(col0 + r) * CH + kb + h;
            *(v8h*)&Bs[r * LDB + h] = *(const v8h*)(s);
        }
        __syncthreads();

        v16h af[2], bf[2];
#pragma unroll
        for (int i = 0; i < 2; ++i) {
            const half_t* ap = &As[(wr + i * 16 + lrow) * LDA + hi * 16];
            v8h alo = *(const v8h*)ap;
            v8h ahi = *(const v8h*)(ap + 8);
            af[i] = __builtin_shufflevector(alo, ahi, 0, 1, 2, 3, 4, 5, 6, 7,
                                            8, 9, 10, 11, 12, 13, 14, 15);
            const half_t* bp = &Bs[(wc + i * 16 + lrow) * LDB + hi * 16];
            v8h blo = *(const v8h*)bp;
            v8h bhi = *(const v8h*)(bp + 8);
            bf[i] = __builtin_shufflevector(blo, bhi, 0, 1, 2, 3, 4, 5, 6, 7,
                                            8, 9, 10, 11, 12, 13, 14, 15);
        }
#pragma unroll
        for (int i = 0; i < 2; ++i)
#pragma unroll
            for (int j = 0; j < 2; ++j)
                acc[i][j] = __builtin_amdgcn_wmma_f32_16x16x32_f16(
                    false, af[i], false, bf[j], (short)0, acc[i][j], false, false);
        __syncthreads();
    }

#pragma unroll
    for (int j = 0; j < 2; ++j) {
        int o = col0 + wc + j * 16 + lrow;
        float bq = bproj[o];
        float g  = bnp[o];
        float be = bnp[CH + o];
        float mn = bnp[2 * CH + o];
        float vr = bnp[3 * CH + o];
        float sc = g * rsqrtf(vr + 1e-5f);
#pragma unroll
        for (int i = 0; i < 2; ++i) {
            int mbase = row0 + wr + i * 16 + hi * 8;
#pragma unroll
            for (int r = 0; r < 8; ++r) {
                int gm = mbase + r;
                int t  = gm / MROWS;
                int rr = gm - t * MROWS;
                int b  = rr / HW;
                int n  = rr - b * HW;
                size_t idx = ((size_t)((t * BATCH + b) * CH + o)) * HW + n;
                float y = (acc[i][j][r] + bq - mn) * sc + be + xid[idx];
                out[idx] = y;
            }
        }
    }
}

// ---------------------------------------------------------------------------
// Second output: v spikes permuted to (T,B,heads,N,hd) as fp32.
// ---------------------------------------------------------------------------
__global__ __launch_bounds__(256) void k_vout(const half_t* __restrict__ Vs,
                                              float* __restrict__ vout) {
    size_t i = (size_t)blockIdx.x * blockDim.x + threadIdx.x;  // 0..MTOT*CH-1
    int d    = (int)(i & 63);
    int rest = (int)(i >> 6);
    int n    = rest % HW;  rest /= HW;
    int head = rest & 7;   rest >>= 3;
    int b    = rest & 31;
    int t    = rest >> 5;
    vout[i] = (float)Vs[((size_t)t * MROWS + b * HW + n) * CH + head * 64 + d];
}

// ---------------------------------------------------------------------------
extern "C" void kernel_launch(void* const* d_in, const int* in_sizes, int n_in,
                              void* d_out, int out_size, void* d_ws, size_t ws_size,
                              hipStream_t stream) {
    (void)in_sizes; (void)n_in; (void)out_size; (void)ws_size;
    const float* x     = (const float*)d_in[0];
    const float* wq    = (const float*)d_in[1];
    const float* wk    = (const float*)d_in[2];
    const float* wv    = (const float*)d_in[3];
    const float* wp    = (const float*)d_in[4];
    const float* bproj = (const float*)d_in[5];
    const float* bn    = (const float*)d_in[6];

    // Workspace layout (f16): Wh[4*C*C] | Xs[MTOT*C] | Qs | Ks | Vs | kvs
    constexpr size_t ACT = (size_t)MTOT * CH;           // 12,845,056 halves
    char*   ws  = (char*)d_ws;
    half_t* Wh  = (half_t*)ws;
    half_t* Xs  = (half_t*)(ws + (size_t)4 * CH * CH * sizeof(half_t));
    half_t* Qs  = Xs + ACT;
    half_t* Ks  = Qs + ACT;
    half_t* Vs  = Ks + ACT;
    half_t* kvs = Vs + ACT;
    half_t* Pin = Xs;  // proj-GEMM input reuses the (dead) Xs buffer

    float* out  = (float*)d_out;
    float* vout = out + ACT;

    k_wcvt<<<(CH * CH) / 256, 256, 0, stream>>>(wq, wk, wv, wp, Wh);
    k_lif_x<<<(MROWS * CH) / 256, 256, 0, stream>>>(x, Xs);

    dim3 gqkv(MROWS / 128, CH / 64, 3);   // 49 x 8 x 3
    k_qkv_gemm_bn_lif<<<gqkv, 256, 0, stream>>>(Xs, Wh, bn, Qs, Ks, Vs);

    k_kv_lif<<<(BATCH * CH) / 256, 256, 0, stream>>>(Ks, Vs, kvs);
    k_mul_q_kv<<<(MTOT * CH + 255) / 256, 256, 0, stream>>>(Qs, kvs, Pin);

    dim3 gproj(MTOT / 128, CH / 64);      // 196 x 8
    k_proj_gemm_bn_add<<<gproj, 256, 0, stream>>>(
        Pin, Wh + (size_t)3 * CH * CH, bproj, bn + 3 * 4 * CH, x, out);

    k_vout<<<(MTOT * CH + 255) / 256, 256, 0, stream>>>(Vs, vout);
}
